// Attention_3487513444997
// MI455X (gfx1250) — matrix operand order repaired and checked by reading, not compiled, not run on hardware
//
#include <hip/hip_runtime.h>
#include <hip/hip_bf16.h>

typedef __attribute__((ext_vector_type(16))) __bf16 v16bf;
typedef __attribute__((ext_vector_type(8)))  __bf16 v8bf;
typedef __attribute__((ext_vector_type(4)))  __bf16 v4bf;
typedef __attribute__((ext_vector_type(8)))  float  v8f;
typedef unsigned int u32;
typedef __attribute__((ext_vector_type(4)))  u32    su4;
typedef __attribute__((ext_vector_type(8)))  u32    su8;

#define BATCH 4
#define SEQ   4096
#define DIM   256
#define KT    32              // keys per tile
#define NT    (SEQ / KT)      // 128 key tiles
#define WAVES 4
#define QPW   16              // queries per wave
#define QPB   (WAVES * QPW)   // 64 queries per block
#define RPW   (KT / WAVES)    // 8 tile rows fetched per wave's TDM slice
#define KROW  272             // ldsK row stride (elements), mult of 8 for 16B-aligned v8bf loads
#define VROW  48              // ldsVt row stride (elements), mult of 16
#define PROW  40              // ldsP row stride (elements), mult of 8 for 16B-aligned v8bf loads

// Issue one Tensor-Data-Mover DMA: 8 rows x 256 fp32 from global -> LDS (contiguous).
// 2-D descriptor, groups 2/3 NULL. Tracked by TENSORcnt.
__device__ __forceinline__ void tdm_load_slice(const float* gsrc, u32 lds_off) {
    const unsigned long long ga = (unsigned long long)(uintptr_t)gsrc;
    su4 g0;
    g0[0] = 1u;                                   // count=1, user descriptor
    g0[1] = lds_off;                              // lds_addr (bytes)
    g0[2] = (u32)ga;                              // global_addr[31:0]  (bits 95:64)
    g0[3] = (u32)(ga >> 32) | 0x80000000u;        // global_addr[56:32] | type=2 (bits 127:126)
    su8 g1;
    g1[0] = 0x00020000u;                          // workgroup_mask=0, data_size=2 (4 bytes)
    g1[1] = (u32)(DIM) << 16;                     // tensor_dim0 = 256 (lo16 at bits 63:48)
    g1[2] = (u32)(RPW) << 16;                     // tensor_dim0 hi=0, tensor_dim1 = 8
    g1[3] = (u32)(DIM) << 16;                     // tensor_dim1 hi=0, tile_dim0 = 256
    g1[4] = (u32)(RPW);                           // tile_dim1 = 8, tile_dim2 = 0
    g1[5] = (u32)(DIM);                           // tensor_dim0_stride = 256 (lo32)
    g1[6] = 0u;                                   // stride hi, tensor_dim1_stride lo
    g1[7] = 0u;
    asm volatile("tensor_load_to_lds %0, %1" :: "s"(g0), "s"(g1) : "memory");
}

__global__ __launch_bounds__(128)
void attn_fa_bf16_wmma(const float* __restrict__ x, float* __restrict__ out) {
    __shared__ float  stage[2][KT * DIM];          // fp32 TDM staging, double-buffered (64KB)
    __shared__ __bf16 ldsK[KT * KROW];             // key tile, row-major [key][dim]
    __shared__ __bf16 ldsVt[DIM * VROW];           // value tile, transposed [dim][key]
    __shared__ __bf16 ldsP[WAVES * QPW * PROW];    // per-wave probability scratch

    const int tid  = threadIdx.x;
    const int lane = tid & 31;
    const int lmod = lane & 15;
    const int lhi  = lane >> 4;
    const int wave = __builtin_amdgcn_readfirstlane(tid >> 5);  // wave-uniform for SGPR descriptors

    const int nqb   = SEQ / QPB;
    const int b     = blockIdx.x / nqb;
    const int qblk  = blockIdx.x % nqb;
    const int qbase = qblk * QPB + wave * QPW;

    const float* xb = x + (size_t)b * SEQ * DIM;

    const u32 stage_off0 = (u32)(uintptr_t)&stage[0][wave * RPW * DIM];
    const u32 stage_off1 = (u32)(uintptr_t)&stage[1][wave * RPW * DIM];

    // ---- preload Q fragments (16x32 bf16 A-matrix layout per 32-dim step) ----
    v16bf qa[8];
    {
        const float* qrow = xb + (size_t)(qbase + lmod) * DIM;
#pragma unroll
        for (int s = 0; s < 8; ++s) {
            const int c1 = s * 32 + lhi * 8;
            const float4 f0 = *(const float4*)(qrow + c1);
            const float4 f1 = *(const float4*)(qrow + c1 + 4);
            const float4 f2 = *(const float4*)(qrow + c1 + 16);
            const float4 f3 = *(const float4*)(qrow + c1 + 20);
            qa[s][0]  = (__bf16)f0.x; qa[s][1]  = (__bf16)f0.y;
            qa[s][2]  = (__bf16)f0.z; qa[s][3]  = (__bf16)f0.w;
            qa[s][4]  = (__bf16)f1.x; qa[s][5]  = (__bf16)f1.y;
            qa[s][6]  = (__bf16)f1.z; qa[s][7]  = (__bf16)f1.w;
            qa[s][8]  = (__bf16)f2.x; qa[s][9]  = (__bf16)f2.y;
            qa[s][10] = (__bf16)f2.z; qa[s][11] = (__bf16)f2.w;
            qa[s][12] = (__bf16)f3.x; qa[s][13] = (__bf16)f3.y;
            qa[s][14] = (__bf16)f3.z; qa[s][15] = (__bf16)f3.w;
        }
    }

    // ---- flash-attention state ----
    v8f acc[16];
#pragma unroll
    for (int t = 0; t < 16; ++t) acc[t] = (v8f){};
    float m_run[8], l_run[8];
#pragma unroll
    for (int r = 0; r < 8; ++r) { m_run[r] = -INFINITY; l_run[r] = 0.0f; }

    __bf16* pbuf = ldsP + wave * QPW * PROW;

    // ---- prologue: async-DMA tile 0 into stage[0] (one 8-row slice per wave) ----
    tdm_load_slice(xb + (size_t)wave * RPW * DIM, stage_off0);

    for (int kt = 0; kt < NT; ++kt) {
        const int cur = kt & 1;

        // wait for this wave's TDM slice of tile kt, then make whole tile visible
        asm volatile("s_wait_tensorcnt 0x0" ::: "memory");
        __syncthreads();

        // ---- convert staged fp32 tile -> bf16 LDS (row-major K + transposed V) ----
        const float* sp = &stage[cur][0];
#pragma unroll
        for (int i = 0; i < 16; ++i) {
            const int idx = i * 512 + tid * 4;
            const int row = idx >> 8;
            const int col = idx & 255;
            const float4 f = *(const float4*)(sp + idx);
            const __bf16 e0 = (__bf16)f.x, e1 = (__bf16)f.y;
            const __bf16 e2 = (__bf16)f.z, e3 = (__bf16)f.w;
            v4bf v; v[0] = e0; v[1] = e1; v[2] = e2; v[3] = e3;
            *(v4bf*)&ldsK[row * KROW + col] = v;           // row-major (K for QK^T)
            ldsVt[(col + 0) * VROW + row] = e0;            // transposed (V for PV)
            ldsVt[(col + 1) * VROW + row] = e1;
            ldsVt[(col + 2) * VROW + row] = e2;
            ldsVt[(col + 3) * VROW + row] = e3;
        }

        // ---- kick off async DMA of the NEXT tile into the other stage buffer ----
        if (kt + 1 < NT)
            tdm_load_slice(xb + ((size_t)(kt + 1) * KT + wave * RPW) * DIM,
                           cur ? stage_off0 : stage_off1);

        __syncthreads();   // bf16 tiles ready for all waves

        // ---- scores: S = Q (16x256) x K^T (256x32), two 16-key subtiles ----
        v8f c0 = (v8f){}, c1 = (v8f){};
#pragma unroll
        for (int s = 0; s < 8; ++s) {
            const int dcol = s * 32 + lhi * 16;
            const v16bf kb0 = __builtin_shufflevector(*(const v8bf*)&ldsK[lmod * KROW + dcol - lhi * 8], *(const v8bf*)&ldsK[lmod * KROW + dcol - lhi * 8 + 16], 0,1,2,3,4,5,6,7,8,9,10,11,12,13,14,15);
            const v16bf kb1 = __builtin_shufflevector(*(const v8bf*)&ldsK[(16 + lmod) * KROW + dcol - lhi * 8], *(const v8bf*)&ldsK[(16 + lmod) * KROW + dcol - lhi * 8 + 16], 0,1,2,3,4,5,6,7,8,9,10,11,12,13,14,15);
            c0 = __builtin_amdgcn_wmma_f32_16x16x32_bf16(false, qa[s], false, kb0,
                                                         (short)0, c0, false, false);
            c1 = __builtin_amdgcn_wmma_f32_16x16x32_bf16(false, qa[s], false, kb1,
                                                         (short)0, c1, false, false);
        }

        // ---- online softmax over this key tile ----
        float scale[8];
#pragma unroll
        for (int r = 0; r < 8; ++r) {
            float mc = fmaxf(c0[r], c1[r]);
            mc = fmaxf(mc, __shfl_xor(mc, 1, 32));
            mc = fmaxf(mc, __shfl_xor(mc, 2, 32));
            mc = fmaxf(mc, __shfl_xor(mc, 4, 32));
            mc = fmaxf(mc, __shfl_xor(mc, 8, 32));
            const float mnew = fmaxf(m_run[r], mc);
            const float sc = __expf(m_run[r] - mnew);
            const float p0 = __expf(c0[r] - mnew);
            const float p1 = __expf(c1[r] - mnew);
            float ps = p0 + p1;
            ps += __shfl_xor(ps, 1, 32);
            ps += __shfl_xor(ps, 2, 32);
            ps += __shfl_xor(ps, 4, 32);
            ps += __shfl_xor(ps, 8, 32);
            l_run[r] = l_run[r] * sc + ps;
            m_run[r] = mnew;
            scale[r] = sc;
            const int prow = (r + 8 * lhi) * PROW;   // C-frag row -> P row-major LDS
            pbuf[prow + lmod]      = (__bf16)p0;
            pbuf[prow + 16 + lmod] = (__bf16)p1;
        }

        // rescale accumulators by exp(m_old - m_new), per row
#pragma unroll
        for (int t = 0; t < 16; ++t)
#pragma unroll
            for (int r = 0; r < 8; ++r)
                acc[t][r] *= scale[r];

        // wave-private LDS round-trip to transpose P into A-fragment layout
        asm volatile("s_wait_dscnt 0x0" ::: "memory");
        v16bf pa;
        {
            const int base = lmod * PROW + lhi * 16;
            const v8bf h0 = *(const v8bf*)(pbuf + base);
            const v8bf h1 = *(const v8bf*)(pbuf + base + 8);
#pragma unroll
            for (int e = 0; e < 8; ++e) { pa[e] = h0[e]; pa[e + 8] = h1[e]; }
        }

        // ---- output: O += P (16x32) x V (32x256), 16 d-tiles ----
#pragma unroll
        for (int t = 0; t < 16; ++t) {
            const v16bf vb = *(const v16bf*)&ldsVt[(t * 16 + lmod) * VROW + lhi * 16];
            acc[t] = __builtin_amdgcn_wmma_f32_16x16x32_bf16(false, pa, false, vb,
                                                             (short)0, acc[t], false, false);
        }
    }

    // ---- finalize: divide by row sums, store fp32 ----
    float inv[8];
#pragma unroll
    for (int r = 0; r < 8; ++r) inv[r] = 1.0f / l_run[r];

    float* orow = out + ((size_t)b * SEQ + qbase) * DIM;
#pragma unroll
    for (int t = 0; t < 16; ++t)
#pragma unroll
        for (int r = 0; r < 8; ++r) {
            const int m = r + 8 * lhi;
            orow[(size_t)m * DIM + t * 16 + lmod] = acc[t][r] * inv[r];
        }
}

extern "C" void kernel_launch(void* const* d_in, const int* in_sizes, int n_in,
                              void* d_out, int out_size, void* d_ws, size_t ws_size,
                              hipStream_t stream) {
    (void)in_sizes; (void)n_in; (void)out_size; (void)d_ws; (void)ws_size;
    const float* x = (const float*)d_in[0];
    float* out = (float*)d_out;
    dim3 grid(BATCH * (SEQ / QPB));   // 4 * 64 = 256 blocks
    attn_fa_bf16_wmma<<<grid, 128, 0, stream>>>(x, out);
}
